// CausalAttention_85048942395739
// MI455X (gfx1250) — compile-verified, matrix operand order checked by reading, not run on hardware
//
#include <hip/hip_runtime.h>
#include <hip/hip_bf16.h>

// Problem constants
#define BB 4
#define SS 2048
#define DD 1024
#define HH 16
#define HD 64

typedef __attribute__((ext_vector_type(16))) __bf16 v16bf;
typedef __attribute__((ext_vector_type(8)))  __bf16 v8bf;
typedef __attribute__((ext_vector_type(8)))  float  v8f;

// ---------------------------------------------------------------------------
// WMMA helper: D = A(16x32 bf16) * B(32x16 bf16) + C(16x16 f32)
// ---------------------------------------------------------------------------
__device__ __forceinline__ v8f wmma_bf16(v16bf a, v16bf b, v8f c) {
  return __builtin_amdgcn_wmma_f32_16x16x32_bf16(
      /*neg_a=*/false, a, /*neg_b=*/false, b,
      /*c_mod=*/(short)0, c, /*reuse_a=*/false, /*reuse_b=*/false);
}

// Load a 16x32 bf16 fragment (A layout; also B layout when the contraction
// dim is contiguous in memory, i.e. source is "transposed"):
//   lane = row(M or N) in [0,16) + 16*k_half; elements 0..7 -> K base..base+7,
//   elements 8..15 -> K base+16..base+23  (base = 8*k_half).
// Two 16B loads per lane.
__device__ __forceinline__ v16bf load_frag(const __bf16* base, int ld, int lane) {
  const int r  = lane & 15;
  const int kh = lane >> 4;
  const __bf16* p = base + (long)r * ld + 8 * kh;
  v8bf lo = *(const v8bf*)(p);
  v8bf hi = *(const v8bf*)(p + 16);
  return __builtin_shufflevector(lo, hi, 0,1,2,3,4,5,6,7,8,9,10,11,12,13,14,15);
}

__device__ __forceinline__ float hmax16(float v) {
  v = fmaxf(v, __shfl_xor(v, 1));
  v = fmaxf(v, __shfl_xor(v, 2));
  v = fmaxf(v, __shfl_xor(v, 4));
  v = fmaxf(v, __shfl_xor(v, 8));
  return v;
}
__device__ __forceinline__ float hsum16(float v) {
  v += __shfl_xor(v, 1);
  v += __shfl_xor(v, 2);
  v += __shfl_xor(v, 4);
  v += __shfl_xor(v, 8);
  return v;
}

// ---------------------------------------------------------------------------
// Elementwise fp32 -> bf16 convert
// ---------------------------------------------------------------------------
__global__ __launch_bounds__(256) void cvt_f32_bf16(const float* __restrict__ in,
                                                    __bf16* __restrict__ out) {
  long i = (long)blockIdx.x * 256 + threadIdx.x;
  out[i] = (__bf16)in[i];
}

// Convert + transpose weights: W[k][n] (D x D, [in,out]) -> Wt[n][k] bf16.
// Pointer selection via if/else so address-space inference keeps global loads.
__global__ __launch_bounds__(256) void cvt_transpose_w(
    const float* __restrict__ Wq, const float* __restrict__ Wk,
    const float* __restrict__ Wv, const float* __restrict__ Wo,
    __bf16* __restrict__ Wqt, __bf16* __restrict__ Wkt,
    __bf16* __restrict__ Wvt, __bf16* __restrict__ Wot) {
  const int z = blockIdx.z;
  const float* src;
  __bf16* dst;
  if (z == 0)      { src = Wq; dst = Wqt; }
  else if (z == 1) { src = Wk; dst = Wkt; }
  else if (z == 2) { src = Wv; dst = Wvt; }
  else             { src = Wo; dst = Wot; }
  long t = (long)blockIdx.x * 256 + threadIdx.x;   // t < D*D
  int k = (int)(t >> 10);
  int n = (int)(t & (DD - 1));
  dst[(long)n * DD + k] = (__bf16)src[t];
}

__global__ __launch_bounds__(256) void zero_f32(float* __restrict__ p) {
  long i = (long)blockIdx.x * 256 + threadIdx.x;
  p[i] = 0.0f;
}

// ---------------------------------------------------------------------------
// QKV projection GEMM: C[M=8192, N=1024] = Xb @ W + bias, per z in {Q,K,V}.
// Workgroup 256 thr = 8 waves arranged 2(M) x 4(N); workgroup tile 64 x 256;
// each wave computes 32(M) x 64(N) = 2x4 WMMA accumulators (8 wmma / k-step).
// Q,K written [B,H,S,HD]; V written transposed [B,H,HD,S] (for P@V B-frags).
// ---------------------------------------------------------------------------
__global__ __launch_bounds__(256) void qkv_gemm(
    const __bf16* __restrict__ Xb,
    const __bf16* __restrict__ Wqt, const __bf16* __restrict__ Wkt,
    const __bf16* __restrict__ Wvt,
    const float* __restrict__ bq, const float* __restrict__ bk,
    const float* __restrict__ bv,
    __bf16* __restrict__ Qh, __bf16* __restrict__ Kh, __bf16* __restrict__ Vt) {
  const int which = blockIdx.z;
  // ternary chains (not private arrays) keep global address space
  const __bf16* Wt   = (which == 0) ? Wqt : (which == 1) ? Wkt : Wvt;
  const float*  bias = (which == 0) ? bq  : (which == 1) ? bk  : bv;

  const int lane = threadIdx.x & 31;
  const int wid  = threadIdx.x >> 5;
  const int wy = wid & 1;           // M direction (2)
  const int wx = wid >> 1;          // N direction (4)
  const int m0 = blockIdx.y * 64 + wy * 32;
  const int n0 = blockIdx.x * 256 + wx * 64;

  const v8f z8 = {0.f,0.f,0.f,0.f,0.f,0.f,0.f,0.f};
  v8f c[2][4];
  #pragma unroll
  for (int i = 0; i < 2; i++)
    #pragma unroll
    for (int j = 0; j < 4; j++) c[i][j] = z8;

  for (int k = 0; k < DD; k += 32) {
    v16bf a0 = load_frag(Xb + (long)m0 * DD + k, DD, lane);
    v16bf a1 = load_frag(Xb + (long)(m0 + 16) * DD + k, DD, lane);
    #pragma unroll
    for (int j = 0; j < 4; j++) {
      v16bf bf = load_frag(Wt + (long)(n0 + j * 16) * DD + k, DD, lane);
      c[0][j] = wmma_bf16(a0, bf, c[0][j]);
      c[1][j] = wmma_bf16(a1, bf, c[1][j]);
    }
  }

  #pragma unroll
  for (int i = 0; i < 2; i++)
    #pragma unroll
    for (int j = 0; j < 4; j++)
      #pragma unroll
      for (int r = 0; r < 8; r++) {
        int sg = m0 + i * 16 + r + 8 * (lane >> 4);   // global row in [0, B*S)
        int n  = n0 + j * 16 + (lane & 15);           // column in [0, D)
        float v = c[i][j][r] + bias[n];
        int b = sg >> 11, s = sg & (SS - 1);
        int h = n >> 6, hd = n & (HD - 1);
        if (which == 0)
          Qh[(((long)b * HH + h) * SS + s) * HD + hd] = (__bf16)v;       // [B,H,S,HD]
        else if (which == 1)
          Kh[(((long)b * HH + h) * SS + s) * HD + hd] = (__bf16)v;       // [B,H,S,HD]
        else
          Vt[(((long)b * HH + h) * HD + hd) * SS + s] = (__bf16)v;       // [B,H,HD,S]
      }
}

// ---------------------------------------------------------------------------
// Attention: one workgroup per (b, h, 64-query tile); 4 waves, 16 q-rows each.
// Two-pass softmax so avg_attention gets normalized probabilities.
// ---------------------------------------------------------------------------
__global__ __launch_bounds__(128) void attn_kernel(
    const __bf16* __restrict__ Qh, const __bf16* __restrict__ Kh,
    const __bf16* __restrict__ Vt,
    __bf16* __restrict__ Ctx, float* __restrict__ avg) {
  __shared__ __align__(16) __bf16 Pst[4][16][32];   // per-wave P staging

  const int lane = threadIdx.x & 31;
  const int wid  = threadIdx.x >> 5;
  const int b = blockIdx.z, h = blockIdx.y;
  const int q0 = blockIdx.x * 64 + wid * 16;

  const __bf16* Qb = Qh + (((long)b * HH + h) * SS + q0) * HD;
  const __bf16* Kb = Kh + (((long)b * HH + h) * SS) * HD;
  const __bf16* Vb = Vt + (((long)b * HH + h) * HD) * SS;

  const v16bf qa0 = load_frag(Qb, HD, lane);
  const v16bf qa1 = load_frag(Qb + 32, HD, lane);
  const float scale = 0.25f;   // 1/(sqrt(64)*0.5)
  const v8f z8 = {0.f,0.f,0.f,0.f,0.f,0.f,0.f,0.f};

  float m[8], l[8];
  #pragma unroll
  for (int r = 0; r < 8; r++) { m[r] = -1e30f; l[r] = 0.0f; }

  // ---------------- pass 1: row max / exp-sum ----------------
  for (int kt = 0; kt < SS; kt += 16) {
    v16bf kb0 = load_frag(Kb + (long)kt * HD, HD, lane);
    v16bf kb1 = load_frag(Kb + (long)kt * HD + 32, HD, lane);
    v8f c = z8;
    c = wmma_bf16(qa0, kb0, c);
    c = wmma_bf16(qa1, kb1, c);
    #pragma unroll
    for (int r = 0; r < 8; r++) {
      float s = c[r] * scale;
      float mx = hmax16(s);
      float mn = fmaxf(m[r], mx);
      float es = hsum16(__expf(s - mn));
      l[r] = l[r] * __expf(m[r] - mn) + es;
      m[r] = mn;
    }
  }
  float inv_l[8];
  #pragma unroll
  for (int r = 0; r < 8; r++) inv_l[r] = 1.0f / l[r];

  // ---------------- pass 2: P, avg, ctx = P @ V ----------------
  v8f ctxacc[4];
  #pragma unroll
  for (int j = 0; j < 4; j++) ctxacc[j] = z8;

  for (int kt = 0; kt < SS; kt += 32) {
    #pragma unroll
    for (int half = 0; half < 2; half++) {
      const int kk = kt + half * 16;
      v16bf kb0 = load_frag(Kb + (long)kk * HD, HD, lane);
      v16bf kb1 = load_frag(Kb + (long)kk * HD + 32, HD, lane);
      v8f c = z8;
      c = wmma_bf16(qa0, kb0, c);
      c = wmma_bf16(qa1, kb1, c);
      #pragma unroll
      for (int r = 0; r < 8; r++) {
        float p = __expf(c[r] * scale - m[r]) * inv_l[r];
        int qrow = q0 + r + 8 * (lane >> 4);
        int kcol = kk + (lane & 15);
        atomicAdd(avg + ((long)b * SS + qrow) * SS + kcol, p * (1.0f / HH));
        Pst[wid][r + 8 * (lane >> 4)][half * 16 + (lane & 15)] = (__bf16)p;
      }
    }
    // same-wave LDS store -> load ordering
    asm volatile("s_wait_dscnt 0" ::: "memory");
    v16bf pa = load_frag(&Pst[wid][0][0], 32, lane);
    #pragma unroll
    for (int j = 0; j < 4; j++) {
      v16bf vb = load_frag(Vb + (long)(j * 16) * SS + kt, SS, lane);
      ctxacc[j] = wmma_bf16(pa, vb, ctxacc[j]);
    }
  }

  // store ctx tile -> [B, S, D] bf16 (D = h*64 + hd)
  #pragma unroll
  for (int j = 0; j < 4; j++)
    #pragma unroll
    for (int r = 0; r < 8; r++) {
      int qrow = q0 + r + 8 * (lane >> 4);
      int d = h * HD + j * 16 + (lane & 15);
      Ctx[((long)b * SS + qrow) * DD + d] = (__bf16)ctxacc[j][r];
    }
}

// ---------------------------------------------------------------------------
// out = LayerNorm(ctx @ Wo + bo + x) * g + beta.
// Workgroup = 16 rows x full D; 8 waves, each wave 16x128 (8 WMMA tiles).
// ---------------------------------------------------------------------------
__global__ __launch_bounds__(256) void out_ln_kernel(
    const __bf16* __restrict__ Ctx, const __bf16* __restrict__ Wot,
    const float* __restrict__ bo, const float* __restrict__ x,
    const float* __restrict__ g, const float* __restrict__ beta,
    float* __restrict__ out) {
  __shared__ float red1[8][16];
  __shared__ float red2[8][16];

  const int lane = threadIdx.x & 31;
  const int wid  = threadIdx.x >> 5;
  const int m0 = blockIdx.x * 16;
  const int n0 = wid * 128;
  const v8f z8 = {0.f,0.f,0.f,0.f,0.f,0.f,0.f,0.f};

  v8f c[8];
  #pragma unroll
  for (int t = 0; t < 8; t++) c[t] = z8;

  for (int k = 0; k < DD; k += 32) {
    v16bf a = load_frag(Ctx + (long)m0 * DD + k, DD, lane);
    #pragma unroll
    for (int t = 0; t < 8; t++) {
      v16bf bfr = load_frag(Wot + (long)(n0 + t * 16) * DD + k, DD, lane);
      c[t] = wmma_bf16(a, bfr, c[t]);
    }
  }

  // h = gemm + bias + residual ; accumulate per-row sum / sumsq
  float s1[8], s2[8];
  #pragma unroll
  for (int r = 0; r < 8; r++) { s1[r] = 0.0f; s2[r] = 0.0f; }
  #pragma unroll
  for (int t = 0; t < 8; t++)
    #pragma unroll
    for (int r = 0; r < 8; r++) {
      int row = m0 + r + 8 * (lane >> 4);
      int n = n0 + t * 16 + (lane & 15);
      float hv = c[t][r] + bo[n] + x[(long)row * DD + n];
      c[t][r] = hv;
      s1[r] += hv;
      s2[r] += hv * hv;
    }
  #pragma unroll
  for (int r = 0; r < 8; r++) { s1[r] = hsum16(s1[r]); s2[r] = hsum16(s2[r]); }

  if ((lane & 15) == 0) {
    #pragma unroll
    for (int r = 0; r < 8; r++) {
      red1[wid][r + 8 * (lane >> 4)] = s1[r];
      red2[wid][r + 8 * (lane >> 4)] = s2[r];
    }
  }
  __syncthreads();

  float mu[8], rs[8];
  #pragma unroll
  for (int r = 0; r < 8; r++) {
    int rl = r + 8 * (lane >> 4);
    float t1 = 0.0f, t2 = 0.0f;
    #pragma unroll
    for (int w = 0; w < 8; w++) { t1 += red1[w][rl]; t2 += red2[w][rl]; }
    float m_ = t1 * (1.0f / DD);
    float v_ = t2 * (1.0f / DD) - m_ * m_;
    mu[r] = m_;
    rs[r] = rsqrtf(v_ + 1e-5f);
  }

  #pragma unroll
  for (int t = 0; t < 8; t++)
    #pragma unroll
    for (int r = 0; r < 8; r++) {
      int row = m0 + r + 8 * (lane >> 4);
      int n = n0 + t * 16 + (lane & 15);
      out[(long)row * DD + n] = (c[t][r] - mu[r]) * rs[r] * g[n] + beta[n];
    }
}

// ---------------------------------------------------------------------------
// Launcher
// ---------------------------------------------------------------------------
extern "C" void kernel_launch(void* const* d_in, const int* in_sizes, int n_in,
                              void* d_out, int out_size, void* d_ws, size_t ws_size,
                              hipStream_t stream) {
  const float* x  = (const float*)d_in[0];
  const float* Wq = (const float*)d_in[1];
  const float* bq = (const float*)d_in[2];
  const float* Wk = (const float*)d_in[3];
  const float* bk = (const float*)d_in[4];
  const float* Wv = (const float*)d_in[5];
  const float* bv = (const float*)d_in[6];
  const float* Wo = (const float*)d_in[7];
  const float* bo = (const float*)d_in[8];
  const float* g  = (const float*)d_in[9];
  const float* be = (const float*)d_in[10];

  // workspace layout (bytes)
  char* ws = (char*)d_ws;
  const size_t SZ_X  = (size_t)BB * SS * DD * 2;      // 16 MB
  const size_t SZ_W  = (size_t)DD * DD * 2;           // 2 MB
  const size_t SZ_QK = (size_t)BB * HH * SS * HD * 2; // 16 MB
  __bf16* xb  = (__bf16*)(ws);
  __bf16* Wqt = (__bf16*)(ws + SZ_X);
  __bf16* Wkt = (__bf16*)(ws + SZ_X + SZ_W);
  __bf16* Wvt = (__bf16*)(ws + SZ_X + 2 * SZ_W);
  __bf16* Wot = (__bf16*)(ws + SZ_X + 3 * SZ_W);
  __bf16* Qh  = (__bf16*)(ws + SZ_X + 4 * SZ_W);
  __bf16* Kh  = (__bf16*)(ws + SZ_X + 4 * SZ_W + SZ_QK);
  __bf16* Vt  = (__bf16*)(ws + SZ_X + 4 * SZ_W + 2 * SZ_QK);
  __bf16* Ctx = (__bf16*)(ws + SZ_X + 4 * SZ_W + 3 * SZ_QK);

  float* out = (float*)d_out;                      // [B,S,D]
  float* avg = out + (long)BB * SS * DD;           // [B,S,S]

  // 1. convert activations + weights to bf16 (weights transposed)
  cvt_f32_bf16<<<(BB * SS * DD) / 256, 256, 0, stream>>>(x, xb);
  cvt_transpose_w<<<dim3(DD * DD / 256, 1, 4), 256, 0, stream>>>(
      Wq, Wk, Wv, Wo, Wqt, Wkt, Wvt, Wot);

  // 2. fused QKV projection (workgroup tile 64x256, wave tile 32x64)
  qkv_gemm<<<dim3(DD / 256, (BB * SS) / 64, 3), 256, 0, stream>>>(
      xb, Wqt, Wkt, Wvt, bq, bk, bv, Qh, Kh, Vt);

  // 3. zero avg_attention (accumulated via atomics)
  zero_f32<<<((long)BB * SS * SS) / 256, 256, 0, stream>>>(avg);

  // 4. attention (two-pass softmax, WMMA QK^T and P@V)
  attn_kernel<<<dim3(SS / 64, HH, BB), 128, 0, stream>>>(Qh, Kh, Vt, Ctx, avg);

  // 5. output projection + residual + LayerNorm
  out_ln_kernel<<<(BB * SS) / 16, 256, 0, stream>>>(Ctx, Wot, bo, x, g, be, out);
}
